// WindowAttention_11338713661474
// MI455X (gfx1250) — compile-verified, matrix operand order checked by reading, not run on hardware
//
#include <hip/hip_runtime.h>
#include <hip/hip_bf16.h>

// ---------------------------------------------------------------------------
// Fused dual-branch window attention for MI455X (gfx1250, wave32, WMMA).
//   prep_kernel    : f32->f16 weight transposes + dense relative-bias table
//   winattn_fused  : per-window QKV GEMM + gated attention + projection,
//                    everything in LDS, all GEMMs via v_wmma_f32_16x16x32_f16
// ---------------------------------------------------------------------------

#define NWIN   2048
#define N_TOK  64
#define C_DIM  192
#define NH     6
#define HD     32
#define C3     576
#define SCALE_Q 0.17677669529663687f   // 32^-0.5

typedef __attribute__((ext_vector_type(16))) _Float16 v16h;
typedef __attribute__((ext_vector_type(8)))  _Float16 v8h;
typedef __attribute__((ext_vector_type(8)))  float    v8f;

// Load a 16x32 f16 WMMA fragment (A from row-major [16 x K], or B from the
// transposed operand, row-major [16(N) x K]).  ld = row pitch in halves.
// lane l (0..31): row = l&15; halves [0..7] = K[ko..ko+7],
// halves [8..15] = K[ko+16..ko+23], ko = (l>>4)*8.  Two 16-byte loads.
__device__ __forceinline__ v16h frag_ld(const _Float16* p, int ld) {
  const int lane = threadIdx.x & 31;
  const int r    = lane & 15;
  const int ko   = (lane >> 4) << 3;
  const _Float16* q = p + r * ld + ko;
  v8h lo = *(const v8h*)(q);
  v8h hi = *(const v8h*)(q + 16);
  return __builtin_shufflevector(lo, hi, 0,1,2,3,4,5,6,7,8,9,10,11,12,13,14,15);
}

__device__ __forceinline__ v8f wmma16(v16h a, v16h b, v8f c) {
  return __builtin_amdgcn_wmma_f32_16x16x32_f16(
      /*neg_a=*/false, a, /*neg_b=*/false, b,
      /*c_mod=*/(short)0, c, /*reuse_a=*/false, /*reuse_b=*/false);
}

// ---------------------------------------------------------------------------
// prep: weight transpose+convert (so WMMA B-fragments are contiguous) and
// dense relative-position bias table  biasg[h][n][m].
// ---------------------------------------------------------------------------
__global__ void prep_kernel(const float* __restrict__ qkv_w,
                            const float* __restrict__ qkv_sar_w,
                            const float* __restrict__ proj_w,
                            const float* __restrict__ proj_sar_w,
                            const float* __restrict__ rel_tab,
                            _Float16* __restrict__ qkv_wt0,
                            _Float16* __restrict__ qkv_wt1,
                            _Float16* __restrict__ proj_wt0,
                            _Float16* __restrict__ proj_wt1,
                            float* __restrict__ biasg) {
  int i = blockIdx.x * blockDim.x + threadIdx.x;
  const int QW = C3 * C_DIM;     // 110592
  const int PW = C_DIM * C_DIM;  // 36864
  if (i < QW) {
    int k = i / C_DIM, c = i - k * C_DIM;
    qkv_wt0[i] = (_Float16)qkv_w[c * C3 + k];
    return;
  }
  i -= QW;
  if (i < QW) {
    int k = i / C_DIM, c = i - k * C_DIM;
    qkv_wt1[i] = (_Float16)qkv_sar_w[c * C3 + k];
    return;
  }
  i -= QW;
  if (i < PW) {
    int k = i / C_DIM, c = i - k * C_DIM;
    proj_wt0[i] = (_Float16)proj_w[c * C_DIM + k];
    return;
  }
  i -= PW;
  if (i < PW) {
    int k = i / C_DIM, c = i - k * C_DIM;
    proj_wt1[i] = (_Float16)proj_sar_w[c * C_DIM + k];
    return;
  }
  i -= PW;
  if (i < NH * N_TOK * N_TOK) {
    int h = i >> 12, nm = i & 4095, n = nm >> 6, m = nm & 63;
    int dh = (n >> 3) - (m >> 3) + 7;   // rows of 8x8 window
    int dw = (n & 7)  - (m & 7)  + 7;
    biasg[i] = rel_tab[(dh * 15 + dw) * NH + h];
  }
}

// ---------------------------------------------------------------------------
// Fused kernel: one block (256 threads = 8 wave32) per window, both branches.
// ---------------------------------------------------------------------------
__launch_bounds__(256)
__global__ void winattn_fused(const float* __restrict__ x,
                              const float* __restrict__ x_sar,
                              const _Float16* __restrict__ qkv_wt0,
                              const _Float16* __restrict__ qkv_wt1,
                              const _Float16* __restrict__ proj_wt0,
                              const _Float16* __restrict__ proj_wt1,
                              const float* __restrict__ biasg,
                              const float* __restrict__ qkv_b,
                              const float* __restrict__ qkv_sar_b,
                              const float* __restrict__ conv_w,
                              const float* __restrict__ conv_b,
                              const float* __restrict__ proj_b,
                              const float* __restrict__ proj_sar_b,
                              float* __restrict__ out) {
  // Persistent LDS: per-head Q (pre-scaled), K row-major, V transposed (f16).
  __shared__ __align__(16) _Float16 QL [2][NH][N_TOK][HD];
  __shared__ __align__(16) _Float16 KL [2][NH][N_TOK][HD];
  __shared__ __align__(16) _Float16 VTL[2][NH][HD][N_TOK];
  // Phase-overlaid LDS: x tiles (QKV phase only) vs logits/probs/attn-out.
  __shared__ __align__(16) union SmemU {
    _Float16 XL[2][N_TOK][C_DIM];                 // 48 KB, dead after QKV
    struct Ph {
      float    S  [2][NH][16][65];                // logits, pitch 65 (banking)
      _Float16 P  [2][NH][16][72];                // probs, pitch 72 (16B rows)
      _Float16 OCF[2][N_TOK][C_DIM];              // full attention output
    } ph;                                         // ~124 KB
  } U;
  __shared__ float qb_s[2][C3];
  __shared__ float pb_s[2][C_DIM];
  __shared__ float cw_s[NH * NH];
  __shared__ float cb_s[NH];

  const int tid  = threadIdx.x;
  const int wid  = tid >> 5;
  const int lane = tid & 31;
  const int col  = lane & 15;
  const int hi8  = (lane >> 4) << 3;
  const int b    = blockIdx.x;

  // ---- preamble: biases / gate weights to LDS, x window to LDS (f16) ----
  for (int i = tid; i < C3; i += 256) {
    qb_s[0][i] = qkv_b[i];
    qb_s[1][i] = qkv_sar_b[i];
  }
  for (int i = tid; i < C_DIM; i += 256) {
    pb_s[0][i] = proj_b[i];
    pb_s[1][i] = proj_sar_b[i];
  }
  if (tid < NH * NH) cw_s[tid] = conv_w[tid];
  if (tid < NH)      cb_s[tid] = conv_b[tid];
  {
    _Float16* xl = &U.XL[0][0][0];
    const size_t base = (size_t)b * (N_TOK * C_DIM);
    for (int i = tid; i < N_TOK * C_DIM; i += 256) {
      xl[i]                 = (_Float16)__builtin_nontemporal_load(x + base + i);
      xl[N_TOK * C_DIM + i] = (_Float16)__builtin_nontemporal_load(x_sar + base + i);
    }
  }
  __syncthreads();

  // ---- QKV GEMM: [64 x 192] x [192 x 576] per branch ----
  // One job = one 16-channel column strip across ALL 4 token tiles, so each
  // global weight fragment is reused 4x in registers.  jobs: br(2)*ct(36)=72.
  for (int j = wid; j < 72; j += 8) {
    const int jj = __builtin_amdgcn_readfirstlane(j);   // wave-uniform decode
    const int br = jj & 1;
    const int ct = jj >> 1;                             // 0..35
    const _Float16* wt = (br ? qkv_wt1 : qkv_wt0) + (size_t)ct * 16 * C_DIM;
    v8f acc[4] = {{}, {}, {}, {}};
#pragma unroll
    for (int ks = 0; ks < 6; ++ks) {
      v16h bfr = frag_ld(wt + ks * 32, C_DIM);
#pragma unroll
      for (int mt = 0; mt < 4; ++mt)
        acc[mt] = wmma16(frag_ld(&U.XL[br][mt * 16][ks * 32], C_DIM), bfr, acc[mt]);
    }
    // Tile-uniform output routing: which = q/k/v, hh = head (scalar).
    const int which = ct / 12;            // 0=q 1=k 2=v
    const int hh    = (ct % 12) >> 1;     // head 0..5
    const int d     = ((ct & 1) << 4) + col;
    const float bia = qb_s[br][ct * 16 + col];
#pragma unroll
    for (int mt = 0; mt < 4; ++mt) {
#pragma unroll
      for (int r = 0; r < 8; ++r) {
        int n = mt * 16 + r + hi8;
        float y = acc[mt][r] + bia;
        if (which == 0)      QL[br][hh][n][d]  = (_Float16)(y * SCALE_Q);
        else if (which == 1) KL[br][hh][n][d]  = (_Float16)y;
        else                 VTL[br][hh][d][n] = (_Float16)y;   // store V^T
      }
    }
  }
  __syncthreads();

  // ---- attention, processed in 4 chunks of 16 query rows ----
  for (int chunk = 0; chunk < 4; ++chunk) {
    // stage 1: S = (Q*scale) K^T   (jobs: br*h*mtile = 48, LDS operands)
    for (int j = wid; j < 48; j += 8) {
      const int jj = __builtin_amdgcn_readfirstlane(j);
      int br = jj & 1, t = jj >> 1, h = t % 6, mt = t / 6;
      v8f sacc = {};
      sacc = wmma16(frag_ld(&QL[br][h][chunk * 16][0], HD),
                    frag_ld(&KL[br][h][mt * 16][0],    HD), sacc);
#pragma unroll
      for (int r = 0; r < 8; ++r)
        U.ph.S[br][h][r + hi8][mt * 16 + col] = sacc[r];
    }
    __syncthreads();

    // stage 2: rel-bias + head-conv sigmoid gate + blend (fp32, elementwise)
    for (int idx = tid; idx < 16 * 64; idx += 256) {
      int nl = idx >> 6, m = idx & 63;
      int n  = chunk * 16 + nl;
      float dv[NH], a0[NH], a1[NH];
#pragma unroll
      for (int h = 0; h < NH; ++h) {
        float s0 = U.ph.S[0][h][nl][m];
        float s1 = U.ph.S[1][h][nl][m];
        float bb = biasg[(h << 12) + (n << 6) + m];
        dv[h] = s1 - s0;           // bias cancels in the difference
        a0[h] = s0 + bb;
        a1[h] = s1 + bb;
      }
#pragma unroll
      for (int g = 0; g < NH; ++g) {
        float z = cb_s[g];
#pragma unroll
        for (int h = 0; h < NH; ++h) z += dv[h] * cw_s[g * NH + h];
        float gate = 1.0f / (1.0f + __expf(-z));
        U.ph.S[0][g][nl][m] = a0[g] + dv[g] * gate;   // blended attn
        U.ph.S[1][g][nl][m] = a1[g];                  // attn_sar
      }
    }
    __syncthreads();

    // stage 3: row softmax (192 rows), write f16 probs
    if (tid < 192) {
      int br = tid / 96, t = tid % 96, h = t >> 4, nl = t & 15;
      const float* row = U.ph.S[br][h][nl];
      float mx = -3.0e38f;
      for (int m = 0; m < 64; ++m) mx = fmaxf(mx, row[m]);
      float sum = 0.0f;
      for (int m = 0; m < 64; ++m) sum += __expf(row[m] - mx);
      float inv = 1.0f / sum;
      for (int m = 0; m < 64; ++m)
        U.ph.P[br][h][nl][m] = (_Float16)(__expf(row[m] - mx) * inv);
    }
    __syncthreads();

    // stage 4: O = P V  (jobs: br*h*dtile = 24, K=64 -> two WMMA steps)
    for (int j = wid; j < 24; j += 8) {
      const int jj = __builtin_amdgcn_readfirstlane(j);
      int br = jj & 1, t = jj >> 1, h = t % 6, dt = t / 6;
      v8f acc = {};
#pragma unroll
      for (int ks = 0; ks < 2; ++ks)
        acc = wmma16(frag_ld(&U.ph.P[br][h][0][ks * 32],       72),
                     frag_ld(&VTL[br][h][dt * 16][ks * 32], N_TOK), acc);
#pragma unroll
      for (int r = 0; r < 8; ++r)
        U.ph.OCF[br][chunk * 16 + r + hi8][h * 32 + dt * 16 + col] = (_Float16)acc[r];
    }
    __syncthreads();
  }

  // ---- projection over the full window: [64 x 192] x [192 x 192] + bias ----
  // One job = 16 output channels across all 4 row tiles (4x weight reuse).
  for (int j = wid; j < 24; j += 8) {
    const int jj = __builtin_amdgcn_readfirstlane(j);
    const int br = jj & 1, nt = jj >> 1;   // nt: 0..11 output-channel tiles
    const _Float16* wt = (br ? proj_wt1 : proj_wt0) + (size_t)nt * 16 * C_DIM;
    v8f acc[4] = {{}, {}, {}, {}};
#pragma unroll
    for (int ks = 0; ks < 6; ++ks) {
      v16h bfr = frag_ld(wt + ks * 32, C_DIM);
#pragma unroll
      for (int mt = 0; mt < 4; ++mt)
        acc[mt] = wmma16(frag_ld(&U.ph.OCF[br][mt * 16][ks * 32], C_DIM), bfr, acc[mt]);
    }
    const int cc = nt * 16 + col;
    const float bia = pb_s[br][cc];
    float* ob = out + (size_t)br * ((size_t)NWIN * N_TOK * C_DIM);
#pragma unroll
    for (int mt = 0; mt < 4; ++mt) {
#pragma unroll
      for (int r = 0; r < 8; ++r) {
        int n = mt * 16 + r + hi8;
        __builtin_nontemporal_store(acc[mt][r] + bia,
            ob + ((size_t)b * N_TOK + n) * C_DIM + cc);
      }
    }
  }
}

// ---------------------------------------------------------------------------
extern "C" void kernel_launch(void* const* d_in, const int* in_sizes, int n_in,
                              void* d_out, int out_size, void* d_ws, size_t ws_size,
                              hipStream_t stream) {
  (void)in_sizes; (void)n_in; (void)out_size; (void)ws_size;
  const float* x          = (const float*)d_in[0];
  const float* x_sar      = (const float*)d_in[1];
  const float* qkv_w      = (const float*)d_in[2];
  const float* qkv_b      = (const float*)d_in[3];
  const float* qkv_sar_w  = (const float*)d_in[4];
  const float* qkv_sar_b  = (const float*)d_in[5];
  const float* rel_tab    = (const float*)d_in[6];
  const float* conv_w     = (const float*)d_in[7];
  const float* conv_b     = (const float*)d_in[8];
  const float* proj_w     = (const float*)d_in[9];
  const float* proj_b     = (const float*)d_in[10];
  const float* proj_sar_w = (const float*)d_in[11];
  const float* proj_sar_b = (const float*)d_in[12];

  // Workspace layout (~690 KB): f16 transposed weights + dense bias table.
  _Float16* qkv_wt0  = (_Float16*)d_ws;             // 576*192 halves
  _Float16* qkv_wt1  = qkv_wt0 + C3 * C_DIM;
  _Float16* proj_wt0 = qkv_wt1 + C3 * C_DIM;        // 192*192 halves
  _Float16* proj_wt1 = proj_wt0 + C_DIM * C_DIM;
  float*    biasg    = (float*)(proj_wt1 + C_DIM * C_DIM);  // 6*64*64 f32

  const int prep_elems = 2 * C3 * C_DIM + 2 * C_DIM * C_DIM + NH * N_TOK * N_TOK;
  prep_kernel<<<(prep_elems + 255) / 256, 256, 0, stream>>>(
      qkv_w, qkv_sar_w, proj_w, proj_sar_w, rel_tab,
      qkv_wt0, qkv_wt1, proj_wt0, proj_wt1, biasg);

  winattn_fused<<<NWIN, 256, 0, stream>>>(
      x, x_sar, qkv_wt0, qkv_wt1, proj_wt0, proj_wt1, biasg,
      qkv_b, qkv_sar_b, conv_w, conv_b, proj_b, proj_sar_b, (float*)d_out);
}
